// Transformer_self_att_11690900980315
// MI455X (gfx1250) — matrix operand order repaired and checked by reading, not compiled, not run on hardware
//
#include <hip/hip_runtime.h>
#include <cstdint>
#include <cstddef>

// ---- CDNA5 wave32 WMMA types -------------------------------------------------
typedef __attribute__((ext_vector_type(16))) __bf16       v16bf;
typedef __attribute__((ext_vector_type(8)))  float        v8f;
typedef __attribute__((ext_vector_type(4)))  unsigned int u32x4;

union FragAB { u32x4 u[2]; v16bf v; };   // 16 bf16 = 32B = two 128-bit chunks

#define TB 256

// ============================================================================
// Positional encoding add + (b,c,h,w) -> (b, l=h*w, c) layout, f32 + bf16 copies
// ============================================================================
__global__ __launch_bounds__(TB) void pos_encode_kernel(
    const float* __restrict__ feats, float* __restrict__ xf, __bf16* __restrict__ xh)
{
  int idx = blockIdx.x * TB + threadIdx.x;
  int ww = idx & 63, hh = (idx >> 6) & 63, c = (idx >> 12) & 255, b = idx >> 20;
  int j = c >> 2, grp = c & 3;
  // div[j] = 10000^(-j/64) = exp(-2j * ln(10000)/128)
  float fdiv = __expf(-(float)(j << 1) * 0.0719557841f);
  float pos  = (grp < 2) ? (float)(ww + 1) : (float)(hh + 1);
  float arg  = pos * fdiv;
  float pe   = (grp & 1) ? __cosf(arg) : __sinf(arg);
  float v = feats[idx] + pe;
  size_t o = ((size_t)b * 4096 + (hh << 6) + ww) * 256 + c;
  xf[o] = v;
  xh[o] = (__bf16)v;
}

// ============================================================================
// Weight repack: W f32 [K][N] -> Wt bf16 [N][K]  (B^T layout for the GEMM)
// ============================================================================
__global__ __launch_bounds__(TB) void repack_wt_kernel(
    const float* __restrict__ W, __bf16* __restrict__ Wt, int K, int N)
{
  int idx = blockIdx.x * TB + threadIdx.x;
  int k = idx / N, n = idx % N;
  Wt[(size_t)n * K + k] = (__bf16)W[idx];
}

// ============================================================================
// bf16 WMMA GEMM: C[M][N] = A[M][K] @ Bt[N][K]^T,  f32 accumulate.
// 256 threads = 8 wave32; block tile 128(M) x 128(N), K-step 32, double-buffered
// LDS (one barrier per K-step, global loads overlap WMMA).
// Waves: wm = wid&3 (4 x 32 rows), wn = wid>>2 (2 x 64 cols); 2x4 WMMA each.
// epi: 0 = f32 store | 1 = ReLU -> bf16 store | 2 = elu(x)+1 on cols<512 -> f32
// ============================================================================
__global__ __launch_bounds__(TB) void gemm_bf16_wmma_kernel(
    const __bf16* __restrict__ A, const __bf16* __restrict__ Bt,
    float* __restrict__ Cf, __bf16* __restrict__ Ch,
    int M, int N, int K, int epi)
{
  __shared__ __bf16 Ah[2][128 * 40];  // padded stride 40 bf16 (80B): no bank conflicts
  __shared__ __bf16 Bh[2][128 * 40];

  const int tid  = threadIdx.x;
  const int lane = tid & 31, wid = tid >> 5;
  const int wm = wid & 3, wn = wid >> 2;    // 4 x 2 wave grid
  const int g = lane >> 4, r = lane & 15;   // lane-group / row-in-tile per ISA layout
  const int ldr = tid >> 2;                 // 0..63 : cooperative load row
  const int ldc = (tid & 3) << 3;           // bf16 col (0,8,16,24) -> one u32x4 each
  const size_t abase = (size_t)(blockIdx.y * 128) * K + ldc;
  const size_t bbase = (size_t)(blockIdx.x * 128) * K + ldc;
  const __bf16* Ap0 = A  + abase + (size_t)ldr        * K;
  const __bf16* Ap1 = A  + abase + (size_t)(ldr + 64) * K;
  const __bf16* Bp0 = Bt + bbase + (size_t)ldr        * K;
  const __bf16* Bp1 = Bt + bbase + (size_t)(ldr + 64) * K;

  v8f acc[2][4] = {};

  // prologue: tile 0 -> LDS buffer 0
  {
    u32x4 ra0 = *(const u32x4*)(Ap0);
    u32x4 ra1 = *(const u32x4*)(Ap1);
    u32x4 rb0 = *(const u32x4*)(Bp0);
    u32x4 rb1 = *(const u32x4*)(Bp1);
    *(u32x4*)&Ah[0][ldr * 40 + ldc]        = ra0;
    *(u32x4*)&Ah[0][(ldr + 64) * 40 + ldc] = ra1;
    *(u32x4*)&Bh[0][ldr * 40 + ldc]        = rb0;
    *(u32x4*)&Bh[0][(ldr + 64) * 40 + ldc] = rb1;
  }
  __syncthreads();

  int buf = 0;
  for (int k0 = 0; k0 < K; k0 += 32, buf ^= 1) {
    const bool more = (k0 + 32) < K;
    u32x4 ra0, ra1, rb0, rb1;
    if (more) {  // stage next K-tile: long-latency global loads issued first
      ra0 = *(const u32x4*)(Ap0 + k0 + 32);
      ra1 = *(const u32x4*)(Ap1 + k0 + 32);
      rb0 = *(const u32x4*)(Bp0 + k0 + 32);
      rb1 = *(const u32x4*)(Bp1 + k0 + 32);
      __builtin_prefetch(Ap0 + k0 + 64, 0, 1);   // -> global_prefetch_b8
    }

    // A fragment (16x32 bf16): lane l, row = r; VGPR0..3 <- K=g*8.. ; VGPR4..7 <- K=16+g*8..
    FragAB a[2], b[4];
    #pragma unroll
    for (int tm = 0; tm < 2; tm++) {
      const __bf16* p = &Ah[buf][(wm * 32 + tm * 16 + r) * 40];
      a[tm].u[0] = *(const u32x4*)(p + g * 8);
      a[tm].u[1] = *(const u32x4*)(p + 16 + g * 8);
    }
    // B fragment (32x16 bf16): lane l, col n = r; VGPR0..3 <- K=g*8.. ; VGPR4..7 <- K=16+g*8.. (same as A)
    #pragma unroll
    for (int tn = 0; tn < 4; tn++) {
      const __bf16* p = &Bh[buf][(wn * 64 + tn * 16 + r) * 40];
      b[tn].u[0] = *(const u32x4*)(p + g * 8);
      b[tn].u[1] = *(const u32x4*)(p + 16 + g * 8);
    }

    if (more) {  // publish next tile into the other buffer
      *(u32x4*)&Ah[buf ^ 1][ldr * 40 + ldc]        = ra0;
      *(u32x4*)&Ah[buf ^ 1][(ldr + 64) * 40 + ldc] = ra1;
      *(u32x4*)&Bh[buf ^ 1][ldr * 40 + ldc]        = rb0;
      *(u32x4*)&Bh[buf ^ 1][(ldr + 64) * 40 + ldc] = rb1;
    }
    __syncthreads();   // single barrier per K-step

    #pragma unroll
    for (int tm = 0; tm < 2; tm++)
      #pragma unroll
      for (int tn = 0; tn < 4; tn++)
        acc[tm][tn] = __builtin_amdgcn_wmma_f32_16x16x32_bf16(
            false, a[tm].v, false, b[tn].v, (short)0, acc[tm][tn], false, false);
  }

  // C/D layout: VGPR rr, lane l -> M = rr + (l>>4)*8, N = l&15
  const int row0 = blockIdx.y * 128 + wm * 32;
  const int col0 = blockIdx.x * 128 + wn * 64;
  for (int tm = 0; tm < 2; tm++)
    for (int tn = 0; tn < 4; tn++) {
      int cc = col0 + tn * 16 + r;
      int rb = row0 + tm * 16 + g * 8;
      #pragma unroll
      for (int rr = 0; rr < 8; rr++) {
        float v = acc[tm][tn][rr];
        size_t o = (size_t)(rb + rr) * N + cc;
        if (epi == 1)      Ch[o] = (__bf16)fmaxf(v, 0.0f);
        else if (epi == 2) Cf[o] = (cc < 512) ? (v > 0.0f ? v + 1.0f : __expf(v)) : v; // elu+1 on q,k
        else               Cf[o] = v;
      }
    }
}

// ============================================================================
// KV[n,h,d,v] = sum_s K[s,d] * V[s,v] ; Ksum[n,h,d] = sum_s K[s,d]
// One block per (n,h); 64-token LDS-staged chunks; thread owns (d, 4 v's).
// ============================================================================
__global__ __launch_bounds__(TB) void kv_kernel(
    const float* __restrict__ qkv, float* __restrict__ KV, float* __restrict__ Ksum)
{
  __shared__ float Ks[64][33];
  __shared__ float Vs[64][33];
  const int t = threadIdx.x;
  const int n = blockIdx.x >> 3, h = blockIdx.x & 7;
  const int d = t >> 3, v0 = (t & 7) << 2;
  float a0 = 0.f, a1 = 0.f, a2 = 0.f, a3 = 0.f, ks = 0.f;
  for (int s0 = 0; s0 < 4096; s0 += 64) {
    for (int i = t; i < 64 * 32; i += TB) {
      int s = i >> 5, dd = i & 31;
      size_t base = ((size_t)(n * 4096 + s0 + s)) * 768 + h * 32 + dd;
      Ks[s][dd] = qkv[base + 256];   // elu(k)+1 (GEMM epilogue)
      Vs[s][dd] = qkv[base + 512];   // raw v (the /L and *L cancel)
    }
    __syncthreads();
    for (int s = 0; s < 64; s++) {
      float kk = Ks[s][d];
      a0 += kk * Vs[s][v0 + 0];
      a1 += kk * Vs[s][v0 + 1];
      a2 += kk * Vs[s][v0 + 2];
      a3 += kk * Vs[s][v0 + 3];
      if ((t & 7) == 0) ks += kk;
    }
    __syncthreads();
  }
  size_t o = ((size_t)blockIdx.x * 32 + d) * 32 + v0;
  KV[o + 0] = a0; KV[o + 1] = a1; KV[o + 2] = a2; KV[o + 3] = a3;
  if ((t & 7) == 0) Ksum[(size_t)blockIdx.x * 32 + d] = ks;
}

// ============================================================================
// attn[l, h*32+v] = (sum_d Q[l,h,d]*KV[h,d,v]) / (Q[l,h,:].Ksum[h,:] + 1e-6)
// One block per 8 tokens; Q/KV/Ksum staged in LDS; bf16 output feeds merge GEMM.
// ============================================================================
__global__ __launch_bounds__(TB) void attn_kernel(
    const float* __restrict__ qkv, const float* __restrict__ KV,
    const float* __restrict__ Ksum, __bf16* __restrict__ attn)
{
  __shared__ float Qs[8][256];
  __shared__ float KVs[8][32][32];
  __shared__ float Kss[256];
  __shared__ float Zs[8][8];
  const int t = threadIdx.x;
  const int l0 = blockIdx.x << 3;
  const int n  = l0 >> 12;
  for (int i = t; i < 8 * 256; i += TB) {
    int li = i >> 8, c = i & 255;
    Qs[li][c] = qkv[(size_t)(l0 + li) * 768 + c];          // elu(q)+1
  }
  for (int i = t; i < 8 * 32 * 32; i += TB)
    ((float*)KVs)[i] = KV[(size_t)n * 8192 + i];
  Kss[t] = Ksum[(size_t)n * 256 + t];
  __syncthreads();
  if (t < 64) {
    int li = t >> 3, h = t & 7;
    float dot = 0.f;
    for (int d2 = 0; d2 < 32; d2++) dot += Qs[li][h * 32 + d2] * Kss[h * 32 + d2];
    Zs[li][h] = 1.0f / (dot + 1e-6f);
  }
  __syncthreads();
  const int h = t >> 5, v = t & 31;
  for (int li = 0; li < 8; li++) {
    float s = 0.f;
    for (int d2 = 0; d2 < 32; d2++) s += Qs[li][h * 32 + d2] * KVs[h][d2][v];
    attn[(size_t)(l0 + li) * 256 + t] = (__bf16)(s * Zs[li][h]);
  }
}

// ============================================================================
// LayerNorm(msg) -> second half of concat (bf16); first half = x (bf16 copy)
// ============================================================================
__global__ __launch_bounds__(TB) void ln1_concat_kernel(
    const float* __restrict__ msg, const __bf16* __restrict__ xh,
    const float* __restrict__ gamma, const float* __restrict__ beta,
    __bf16* __restrict__ cat)
{
  __shared__ float red[TB];
  const int l = blockIdx.x, t = threadIdx.x;
  float v = msg[(size_t)l * 256 + t];
  red[t] = v; __syncthreads();
  for (int s = 128; s > 0; s >>= 1) { if (t < s) red[t] += red[t + s]; __syncthreads(); }
  float mu = red[0] * (1.0f / 256.0f);
  __syncthreads();
  float c = v - mu;
  red[t] = c * c; __syncthreads();
  for (int s = 128; s > 0; s >>= 1) { if (t < s) red[t] += red[t + s]; __syncthreads(); }
  float var = red[0] * (1.0f / 256.0f);
  float y = c * rsqrtf(var + 1e-5f) * gamma[t] + beta[t];
  cat[(size_t)l * 512 + t]       = xh[(size_t)l * 256 + t];
  cat[(size_t)l * 512 + 256 + t] = (__bf16)y;
}

// ============================================================================
// x = x + LayerNorm(mlp_out); refresh f32 state and bf16 copy
// ============================================================================
__global__ __launch_bounds__(TB) void ln2_res_kernel(
    const float* __restrict__ mlp, const float* __restrict__ gamma,
    const float* __restrict__ beta, float* __restrict__ xf, __bf16* __restrict__ xh)
{
  __shared__ float red[TB];
  const int l = blockIdx.x, t = threadIdx.x;
  float v = mlp[(size_t)l * 256 + t];
  red[t] = v; __syncthreads();
  for (int s = 128; s > 0; s >>= 1) { if (t < s) red[t] += red[t + s]; __syncthreads(); }
  float mu = red[0] * (1.0f / 256.0f);
  __syncthreads();
  float c = v - mu;
  red[t] = c * c; __syncthreads();
  for (int s = 128; s > 0; s >>= 1) { if (t < s) red[t] += red[t + s]; __syncthreads(); }
  float var = red[0] * (1.0f / 256.0f);
  float y  = c * rsqrtf(var + 1e-5f) * gamma[t] + beta[t];
  float xn = xf[(size_t)l * 256 + t] + y;
  xf[(size_t)l * 256 + t] = xn;
  xh[(size_t)l * 256 + t] = (__bf16)xn;
}

// ============================================================================
// (b, l, c) f32 -> (b, c, h, w) f32 output
// ============================================================================
__global__ __launch_bounds__(TB) void output_kernel(
    const float* __restrict__ xf, float* __restrict__ out)
{
  int idx = blockIdx.x * TB + threadIdx.x;
  int ww = idx & 63, hh = (idx >> 6) & 63, c = (idx >> 12) & 255, b = idx >> 20;
  out[idx] = xf[((size_t)b * 4096 + (hh << 6) + ww) * 256 + c];
}

// ============================================================================
extern "C" void kernel_launch(void* const* d_in, const int* in_sizes, int n_in,
                              void* d_out, int out_size, void* d_ws, size_t ws_size,
                              hipStream_t stream)
{
  (void)in_sizes; (void)n_in; (void)out_size; (void)ws_size;
  const float* feats = (const float*)d_in[0];
  const float* Wq = (const float*)d_in[1];
  const float* Wk = (const float*)d_in[2];
  const float* Wv = (const float*)d_in[3];
  const float* Wm = (const float*)d_in[4];
  const float* W1 = (const float*)d_in[5];
  const float* W2 = (const float*)d_in[6];
  const float* g1 = (const float*)d_in[7];
  const float* b1 = (const float*)d_in[8];
  const float* g2 = (const float*)d_in[9];
  const float* b2 = (const float*)d_in[10];
  float* out = (float*)d_out;

  const int M = 32768;                       // 8 batches * 4096 tokens
  char* w = (char*)d_ws;
  size_t off = 0;
  auto carve = [&](size_t bytes) -> char* {
    char* p = w + off; off = (off + bytes + 255) & ~(size_t)255; return p;
  };

  float*  xf   = (float*) carve((size_t)M * 256 * 4);   // residual state (f32)
  __bf16* xh   = (__bf16*)carve((size_t)M * 256 * 2);   // bf16 copy for GEMM A
  float*  qkv  = (float*) carve((size_t)M * 768 * 4);   // [q|k|v] post-activation
  float*  msg  = qkv;                                   // alias (qkv dead after attn)
  __bf16* hid  = (__bf16*)((char*)qkv + (size_t)M * 256 * 4);
  float*  mlp  = (float*) ((char*)qkv + (size_t)M * 512 * 4);
  __bf16* attn = (__bf16*)carve((size_t)M * 256 * 2);
  __bf16* cat  = (__bf16*)carve((size_t)M * 512 * 2);
  float*  KVb  = (float*) carve(8 * 8 * 32 * 32 * 4);
  float*  Ksum = (float*) carve(8 * 8 * 32 * 4);
  __bf16* wqkv = (__bf16*)carve(768 * 256 * 2);         // [Wq^T | Wk^T | Wv^T]
  __bf16* wmt  = (__bf16*)carve(256 * 256 * 2);
  __bf16* w1t  = (__bf16*)carve(512 * 512 * 2);
  __bf16* w2t  = (__bf16*)carve(256 * 512 * 2);

  pos_encode_kernel<<<(8 * 256 * 64 * 64) / TB, TB, 0, stream>>>(feats, xf, xh);

  for (int i = 0; i < 4; i++) {
    repack_wt_kernel<<<(256 * 256) / TB, TB, 0, stream>>>(Wq + i * 65536,  wqkv,          256, 256);
    repack_wt_kernel<<<(256 * 256) / TB, TB, 0, stream>>>(Wk + i * 65536,  wqkv + 65536,  256, 256);
    repack_wt_kernel<<<(256 * 256) / TB, TB, 0, stream>>>(Wv + i * 65536,  wqkv + 131072, 256, 256);
    repack_wt_kernel<<<(256 * 256) / TB, TB, 0, stream>>>(Wm + i * 65536,  wmt, 256, 256);
    repack_wt_kernel<<<(512 * 512) / TB, TB, 0, stream>>>(W1 + i * 262144, w1t, 512, 512);
    repack_wt_kernel<<<(512 * 256) / TB, TB, 0, stream>>>(W2 + i * 131072, w2t, 512, 256);

    // fused QKV projection, elu+1 on q,k columns in epilogue
    gemm_bf16_wmma_kernel<<<dim3(768 / 128, M / 128), TB, 0, stream>>>(
        xh, wqkv, qkv, nullptr, M, 768, 256, 2);
    kv_kernel<<<64, TB, 0, stream>>>(qkv, KVb, Ksum);
    attn_kernel<<<M / 8, TB, 0, stream>>>(qkv, KVb, Ksum, attn);
    // merge projection
    gemm_bf16_wmma_kernel<<<dim3(256 / 128, M / 128), TB, 0, stream>>>(
        attn, wmt, msg, nullptr, M, 256, 256, 0);
    ln1_concat_kernel<<<M, TB, 0, stream>>>(msg, xh, g1 + i * 256, b1 + i * 256, cat);
    // MLP1 with ReLU->bf16 epilogue, MLP2
    gemm_bf16_wmma_kernel<<<dim3(512 / 128, M / 128), TB, 0, stream>>>(
        cat, w1t, nullptr, hid, M, 512, 512, 1);
    gemm_bf16_wmma_kernel<<<dim3(256 / 128, M / 128), TB, 0, stream>>>(
        hid, w2t, mlp, nullptr, M, 256, 512, 0);
    ln2_res_kernel<<<M, TB, 0, stream>>>(mlp, g2 + i * 256, b2 + i * 256, xf, xh);
  }

  output_kernel<<<(8 * 256 * 64 * 64) / TB, TB, 0, stream>>>(xf, out);
}
